// GCNLayer_30193620090942
// MI455X (gfx1250) — matrix operand order repaired and checked by reading, not compiled, not run on hardware
//
#include <hip/hip_runtime.h>

#define N_NODES 100000
#define N_EDGES 600000
#define D 128

typedef __attribute__((ext_vector_type(16))) _Float16 v16h;
typedef __attribute__((ext_vector_type(8)))  _Float16 v8h;
typedef __attribute__((ext_vector_type(8)))  float    v8f;

// ---------------- zero init (deg + agg) ----------------
__global__ __launch_bounds__(256) void k_zero(float* deg, float* agg) {
    long long idx    = (long long)blockIdx.x * blockDim.x + threadIdx.x;
    long long stride = (long long)gridDim.x * blockDim.x;
    const long long total = (long long)N_NODES * D;
    for (long long i = idx; i < total; i += stride) agg[i] = 0.0f;
    for (long long i = idx; i < N_NODES; i += stride) deg[i] = 0.0f;
}

// ---------------- in-degree via f32 atomics ----------------
__global__ __launch_bounds__(256) void k_degree(const int* __restrict__ dst,
                                                float* __restrict__ deg) {
    int e = blockIdx.x * blockDim.x + threadIdx.x;
    if (e < N_EDGES) atomicAdd(&deg[dst[e]], 1.0f);
}

// ---------------- norm = rsqrt(max(deg,1)) in place ----------------
__global__ __launch_bounds__(256) void k_norm(float* __restrict__ deg) {
    int i = blockIdx.x * blockDim.x + threadIdx.x;
    if (i < N_NODES) {
        float d = deg[i];
        deg[i] = rsqrtf(d < 1.0f ? 1.0f : d);
    }
}

// ---------------- edge scatter: agg[dst] += feat[src]*norm[src] ----------------
// one wave32 per edge; each lane handles 4 consecutive floats (float4 load + 4 atomics)
__global__ __launch_bounds__(256) void k_scatter(const int* __restrict__ src,
                                                 const int* __restrict__ dst,
                                                 const float* __restrict__ feat,
                                                 const float* __restrict__ norm,
                                                 float* __restrict__ agg) {
    int wave = (int)(((long long)blockIdx.x * blockDim.x + threadIdx.x) >> 5);
    int lane = threadIdx.x & 31;
    if (wave >= N_EDGES) return;
    int s = src[wave];
    int d = dst[wave];
    float ns = norm[s];
    const float4* fs = (const float4*)(feat + (long long)s * D);
    float4 v = fs[lane];
    float* ad = agg + (long long)d * D + lane * 4;
    atomicAdd(ad + 0, v.x * ns);
    atomicAdd(ad + 1, v.y * ns);
    atomicAdd(ad + 2, v.z * ns);
    atomicAdd(ad + 3, v.w * ns);
}

// ---------------- post-normalize + convert f32 -> f16 (A operand) ----------------
__global__ __launch_bounds__(256) void k_convA(const float* __restrict__ agg,
                                               const float* __restrict__ norm,
                                               _Float16* __restrict__ hA) {
    long long i = (long long)blockIdx.x * blockDim.x + threadIdx.x;
    if (i < (long long)N_NODES * D) {
        int row = (int)(i >> 7);
        hA[i] = (_Float16)(agg[i] * norm[row]);
    }
}

// ---------------- W [k][n] -> Wt [n][k] in f16 (B operand, column-major) ----------------
__global__ __launch_bounds__(256) void k_convW(const float* __restrict__ W,
                                               _Float16* __restrict__ Wt) {
    int i = blockIdx.x * blockDim.x + threadIdx.x;
    if (i < D * D) {
        int k = i >> 7;
        int n = i & 127;
        Wt[n * D + k] = (_Float16)W[i];
    }
}

// ---------------- WMMA GEMM: out = hA @ W + bias ----------------
// one wave per 16x16 output tile; 6250 blocks x 256 threads = 50000 waves = 6250x8 tiles
__global__ __launch_bounds__(256) void k_gemm(const _Float16* __restrict__ hA,
                                              const _Float16* __restrict__ Wt,
                                              const float* __restrict__ bias,
                                              float* __restrict__ out) {
    const int lane  = threadIdx.x & 31;
    const int wave  = (int)(((long long)blockIdx.x * blockDim.x + threadIdx.x) >> 5);
    const int mtile = wave >> 3;      // 0..6249
    const int ntile = wave & 7;       // 0..7
    const int l15   = lane & 15;
    const int hi    = lane >> 4;      // 0 or 1 (lane half)

    // A (16-bit 16x32 layout): lane l<16 holds M=l, halves {K0..7, K16..23};
    // lanes 16..31 hold M=l-16, halves {K8..15, K24..31}.
    const _Float16* Arow = hA + (long long)(mtile * 16 + l15) * D;
    // B (16-bit 32x16 layout): lane l<16 holds N=l, K {0..7, 16..23}; lanes 16..31 K {8..15, 24..31}.
    const _Float16* Bcol = Wt + (long long)(ntile * 16 + l15) * D + hi * 8;

    v8f c = {};
#pragma unroll
    for (int kk = 0; kk < 4; ++kk) {
        const int k0 = kk * 32;
        v8h a_lo = *(const v8h*)(Arow + k0 + hi * 8);        // halves 0..7
        v8h a_hi = *(const v8h*)(Arow + k0 + hi * 8 + 16);   // halves 8..15
        v8h b_lo = *(const v8h*)(Bcol + k0);                 // halves 0..7
        v8h b_hi = *(const v8h*)(Bcol + k0 + 16);
        v16h a, b;
#pragma unroll
        for (int t = 0; t < 8; ++t) {
            a[t] = a_lo[t]; a[t + 8] = a_hi[t];
            b[t] = b_lo[t]; b[t + 8] = b_hi[t];
        }
        // 8 args: (neg_a, A, neg_b, B, c_mod, C, reuse_a, reuse_b)
        c = __builtin_amdgcn_wmma_f32_16x16x32_f16(false, a, false, b,
                                                   (short)0, c, false, false);
    }

    // C/D layout: VGPR r -> row M = r + hi*8, col N = l15
    const float bval = bias[ntile * 16 + l15];
    const int rowbase = mtile * 16 + hi * 8;
#pragma unroll
    for (int r = 0; r < 8; ++r) {
        out[(long long)(rowbase + r) * D + ntile * 16 + l15] = c[r] + bval;
    }
}

extern "C" void kernel_launch(void* const* d_in, const int* in_sizes, int n_in,
                              void* d_out, int out_size, void* d_ws, size_t ws_size,
                              hipStream_t stream) {
    const float* feat   = (const float*)d_in[0];
    const float* weight = (const float*)d_in[1];
    const float* bias   = (const float*)d_in[2];
    const int*   src    = (const int*)d_in[3];
    const int*   dst    = (const int*)d_in[4];
    float*       out    = (float*)d_out;

    // workspace layout (all 256B-aligned offsets)
    char* ws = (char*)d_ws;
    float*    norm = (float*)ws;                                   // 400 KB
    float*    agg  = (float*)(ws + (size_t)512 * 1024);            // 51.2 MB
    _Float16* hA   = (_Float16*)(ws + (size_t)512 * 1024
                                    + (size_t)N_NODES * D * 4);    // 25.6 MB
    _Float16* Wt   = (_Float16*)(ws + (size_t)512 * 1024
                                    + (size_t)N_NODES * D * 4
                                    + (size_t)N_NODES * D * 2);    // 32 KB

    k_zero<<<4096, 256, 0, stream>>>(norm, agg);
    k_degree<<<(N_EDGES + 255) / 256, 256, 0, stream>>>(dst, norm);
    k_norm<<<(N_NODES + 255) / 256, 256, 0, stream>>>(norm);
    // one wave per edge: 600000 waves / 8 waves-per-block = 75000 blocks
    k_scatter<<<N_EDGES / 8, 256, 0, stream>>>(src, dst, feat, norm, agg);
    k_convA<<<(N_NODES * D) / 256, 256, 0, stream>>>(agg, norm, hA);
    k_convW<<<(D * D) / 256, 256, 0, stream>>>(weight, Wt);
    // 6250 blocks x 8 waves = 50000 waves = (100000/16) x (128/16) tiles
    k_gemm<<<N_NODES / 16, 256, 0, stream>>>(hA, Wt, bias, out);
}
